// CausalSelfAttentionRelPos_25701084299441
// MI455X (gfx1250) — matrix operand order repaired and checked by reading, not compiled, not run on hardware
//
#include <hip/hip_runtime.h>
#include <cstdint>

typedef unsigned short u16;
typedef __attribute__((ext_vector_type(16))) __bf16 v16bf;
typedef __attribute__((ext_vector_type(8)))  float  v8f;
typedef __attribute__((ext_vector_type(4)))  unsigned int u32x4;
typedef __attribute__((ext_vector_type(8)))  int i32x8;
typedef __attribute__((ext_vector_type(4)))  int i32x4;

union FragA { v16bf v; uint4 u[2]; };

__device__ __forceinline__ u16 f2bf(float f){
  union { float f; unsigned int i; } w; w.f = f;
  unsigned int r = (w.i + 0x7FFFu + ((w.i >> 16) & 1u)) >> 16;
  return (u16)r;
}
__device__ __forceinline__ float bf2f(u16 u){
  union { float f; unsigned int i; } w; w.i = ((unsigned int)u) << 16; return w.f;
}

__device__ __forceinline__ v8f wmma_bf16(v16bf a, v16bf b, v8f c){
  // (neg_a, A, neg_b, B, c_mod, C, reuse_a, reuse_b)
  return __builtin_amdgcn_wmma_f32_16x16x32_bf16(false, a, false, b, (short)0, c, false, false);
}

__device__ __forceinline__ float rmax16(float x){
  x = fmaxf(x, __shfl_xor(x, 1, 32));
  x = fmaxf(x, __shfl_xor(x, 2, 32));
  x = fmaxf(x, __shfl_xor(x, 4, 32));
  x = fmaxf(x, __shfl_xor(x, 8, 32));
  return x;
}
__device__ __forceinline__ float rsum16(float x){
  x += __shfl_xor(x, 1, 32);
  x += __shfl_xor(x, 2, 32);
  x += __shfl_xor(x, 4, 32);
  x += __shfl_xor(x, 8, 32);
  return x;
}

// async global->LDS copy of one 16B chunk (per-lane addresses), tracked by ASYNCcnt
__device__ __forceinline__ void async_copy_b128(unsigned lds_off, const u16* gaddr){
  asm volatile("global_load_async_to_lds_b128 %0, %1, off"
               :: "v"(lds_off), "v"(gaddr) : "memory");
}
__device__ __forceinline__ void wait_async0(){
  asm volatile("s_wait_asynccnt 0" ::: "memory");
}

// TDM: load a 128x32 bf16 2D tile (row stride `stride` elems) into LDS with
// row padding 32 data elems + 8 pad elems (pitch 40). Tracked by TENSORcnt.
__device__ __forceinline__ void tdm_load_tile(unsigned lds_off, const u16* gaddr,
                                              int stride, int tens_rows){
  unsigned long long ga = (unsigned long long)(uintptr_t)gaddr;
  u32x4 g0 = {0u, 0u, 0u, 0u};
  g0[0] = 1u;                                        // count=1, user descriptor
  g0[1] = lds_off;                                   // lds_addr (bytes)
  g0[2] = (unsigned)(ga & 0xFFFFFFFFu);              // global_addr[31:0]
  g0[3] = (unsigned)((ga >> 32) & 0x01FFFFFFu) | (2u << 30);  // addr[56:32], type=2
  i32x8 g1 = {0,0,0,0,0,0,0,0};
  // data_size=1 (2B), pad_enable, pad_interval=3 (16 DWORDs), pad_amount=3 (4 DWORDs)
  g1[0] = (1 << 16) | (1 << 20) | (3 << 22) | (3 << 25);
  g1[1] = (int)((1024u & 0xFFFFu) << 16);            // tensor_dim0[15:0] = 1024
  g1[2] = (int)(((1024u >> 16) & 0xFFFFu) |
                (((unsigned)tens_rows & 0xFFFFu) << 16));     // dim0 hi | dim1 lo
  g1[3] = (int)((((unsigned)tens_rows >> 16) & 0xFFFFu) |
                (32u << 16));                        // dim1 hi | tile_dim0=32
  g1[4] = (int)(128u);                               // tile_dim1=128, tile_dim2=0
  g1[5] = stride;                                    // tensor_dim0_stride[31:0]
  i32x4 g2 = {0,0,0,0};                              // 2D tile: dims 2..4 unused
  i32x4 g3 = {0,0,0,0};
  i32x8 g4 = {0,0,0,0,0,0,0,0};                      // extra group (6-arg toolchain form)
  __builtin_amdgcn_tensor_load_to_lds(g0, g1, g2, g3, g4, 0);
}

// ---------------- converters ----------------

__global__ __launch_bounds__(256) void cvt_bf16_kernel(const float* __restrict__ in,
                                                       u16* __restrict__ out, int n){
  int i = blockIdx.x * 256 + threadIdx.x;
  if (i < n) out[i] = f2bf(in[i]);
}

// in[R][C] f32 -> out[C][R] bf16, LDS-tiled 32x32
__global__ __launch_bounds__(256) void transpose_f32_bf16(const float* __restrict__ in,
                                                          u16* __restrict__ out, int R, int C){
  __shared__ u16 tile[32][33];
  int tx = threadIdx.x & 31, ty = threadIdx.x >> 5;  // 32 x 8
  int r0 = blockIdx.y * 32, c0 = blockIdx.x * 32;
  #pragma unroll
  for (int i = 0; i < 32; i += 8)
    tile[ty + i][tx] = f2bf(in[(size_t)(r0 + ty + i) * C + c0 + tx]);
  __syncthreads();
  #pragma unroll
  for (int i = 0; i < 32; i += 8)
    out[(size_t)(c0 + ty + i) * R + r0 + tx] = tile[tx][ty + i];
}

// ---------------- WMMA GEMM: C[128x128] per block, K=1024, bf16 inputs ----------------
// Double-buffered staging: A tile via ASYNCcnt global->LDS b128 copies,
// B tile via one TDM tensor_load_to_lds descriptor per K-step (TENSORcnt).
// 8 waves x (4x2) 16x16x32 WMMA accumulators, manual 2x unrolled pipeline.
// mode 0: QKV epilogue (bf16 q,k token-major; v transposed [bh][d][l]) + b_attn
// mode 1: f32 out + b_proj

__global__ __launch_bounds__(256) void gemm_bf16(const u16* __restrict__ Ag, int lda,
    const u16* __restrict__ Btg, int nB, const float* __restrict__ bias, int mode,
    u16* __restrict__ qb, u16* __restrict__ kb, u16* __restrict__ vtb,
    float* __restrict__ outf)
{
  // 2 buffers x (A 128x40 + B 128x40) u16 = 40960 B; epilogue reuses first 34816 B
  __shared__ u16 smem[20480];
  const int tid  = threadIdx.x;
  const int lane = tid & 31, wave = tid >> 5;
  const int wr = wave >> 2, wc = wave & 3;       // 2 x 4 wave grid
  const int half = lane >> 4, colL = lane & 15;
  const int m0 = blockIdx.x << 7, n0 = blockIdx.y << 7;
  const v8f vz = {0.f,0.f,0.f,0.f,0.f,0.f,0.f,0.f};
  v8f acc[4][2];
  #pragma unroll
  for (int a = 0; a < 4; ++a)
    #pragma unroll
    for (int b = 0; b < 2; ++b) acc[a][b] = vz;

  // A tile: 512 x 16B chunks, 2 per thread, async to LDS
  auto stageA = [&](int buf, int k0){
    #pragma unroll
    for (int it = 0; it < 2; ++it){
      int cc = tid + (it << 8);
      int row = cc >> 2, part = cc & 3;
      unsigned lA = (unsigned)(uintptr_t)&smem[buf*10240 + row*40 + part*8];
      async_copy_b128(lA, &Ag[(size_t)(m0+row)*lda + k0 + part*8]);
    }
  };
  // B tile: single TDM descriptor issued by wave 0
  auto stageB = [&](int buf, int k0){
    if (wave == 0)
      tdm_load_tile((unsigned)(uintptr_t)&smem[buf*10240 + 5120],
                    &Btg[(size_t)n0*1024 + k0], 1024, nB);
  };
  auto settle = [&](){
    wait_async0();
    if (wave == 0) __builtin_amdgcn_s_wait_tensorcnt(0);
    __syncthreads();
  };
  auto compute = [&](int buf){
    const u16* At  = &smem[buf*10240];
    const u16* Btl = &smem[buf*10240 + 5120];
    FragA af[4], bfr[2];
    #pragma unroll
    for (int mt = 0; mt < 4; ++mt){
      const u16* p = &At[(wr*64 + mt*16 + colL)*40 + half*16];
      af[mt].u[0] = *(const uint4*)p;
      af[mt].u[1] = *(const uint4*)(p + 8);
    }
    #pragma unroll
    for (int nt = 0; nt < 2; ++nt){
      const u16* p = &Btl[(wc*32 + nt*16 + colL)*40 + half*16];
      bfr[nt].u[0] = *(const uint4*)p;
      bfr[nt].u[1] = *(const uint4*)(p + 8);
    }
    #pragma unroll
    for (int mt = 0; mt < 4; ++mt)
      #pragma unroll
      for (int nt = 0; nt < 2; ++nt)
        acc[mt][nt] = wmma_bf16(af[mt].v, bfr[nt].v, acc[mt][nt]);
  };

  stageB(0, 0);
  stageA(0, 0);
  settle();
  for (int k0 = 0; k0 < 1024; k0 += 64){
    if (k0 + 32 < 1024){ stageB(1, k0 + 32); stageA(1, k0 + 32); }
    compute(0);
    settle();
    if (k0 + 64 < 1024){ stageB(0, k0 + 64); stageA(0, k0 + 64); }
    compute(1);
    settle();
  }

  // epilogue through LDS (pitch 136)
  #pragma unroll
  for (int mt = 0; mt < 4; ++mt)
    #pragma unroll
    for (int nt = 0; nt < 2; ++nt)
      #pragma unroll
      for (int i = 0; i < 8; ++i){
        int r = wr*64 + mt*16 + i + half*8;
        int c = wc*32 + nt*16 + colL;
        smem[r*136 + c] = f2bf(acc[mt][nt][i]);
      }
  __syncthreads();

  if (mode == 0){
    for (int it = 0; it < 64; ++it){
      int idx = tid + (it << 8);
      int r = idx >> 7, c = idx & 127;
      int n = n0 + c;
      float val = bf2f(smem[r*136 + c]) + bias[n];
      int m = m0 + r;
      int bb = m >> 11, l = m & 2047;
      int mat = n >> 10, rem = n & 1023;
      int h = rem >> 6, dd = rem & 63;
      size_t bh = (size_t)bb*16 + h;
      u16 bv = f2bf(val);
      if (mat == 0)      qb [(bh*2048 + l)*64 + dd] = bv;
      else if (mat == 1) kb [(bh*2048 + l)*64 + dd] = bv;
      else               vtb[(bh*64 + dd)*2048 + l] = bv;
    }
  } else {
    for (int it = 0; it < 64; ++it){
      int idx = tid + (it << 8);
      int r = idx >> 7, c = idx & 127;
      int n = n0 + c;
      outf[(size_t)(m0 + r)*1024 + n] = bf2f(smem[r*136 + c]) + bias[n];
    }
  }
}

// ---------------- Flash attention with rel-pos skew ----------------
// 1 wave == one 16-row query tile of one (b,h). 4 independent waves per block.

#define LQ 2048

__global__ __launch_bounds__(128) void attn_kernel(const u16* __restrict__ q,
    const u16* __restrict__ k, const u16* __restrict__ vt,
    const u16* __restrict__ er, u16* __restrict__ yatt)
{
  __shared__ u16 pall[4*16*40];                  // per-wave P transpose scratch
  const int lane = threadIdx.x & 31;
  const int wave = threadIdx.x >> 5;
  u16* pscr = pall + wave*16*40;
  const int tile = blockIdx.x*4 + wave;
  const int bh = tile >> 7;                      // (b*16+h)
  const int l0 = (tile & 127) << 4;
  const int half = lane >> 4, colL = lane & 15;
  const float scale = 0.125f;                    // 1/sqrt(64)
  const v8f vz = {0.f,0.f,0.f,0.f,0.f,0.f,0.f,0.f};

  // Q A-fragments (rows l0..l0+15, K halves d=0..31 / 32..63)
  FragA aq[2];
  const u16* qrow = q + ((size_t)bh*LQ + l0 + colL)*64;
  #pragma unroll
  for (int f = 0; f < 2; ++f){
    const u16* p = qrow + 32*f + half*16;
    aq[f].u[0] = *(const uint4*)p;
    aq[f].u[1] = *(const uint4*)(p + 8);
  }

  v8f o[4];
  #pragma unroll
  for (int i = 0; i < 4; ++i) o[i] = vz;
  float m_i[8], l_i[8];
  #pragma unroll
  for (int i = 0; i < 8; ++i){ m_i[i] = -1e30f; l_i[i] = 0.f; }

  const int cs = LQ - 16 - l0;                   // G band origin: c = cs + 15 + mi - li + m0

  auto gtile = [&](int c0) -> v8f {              // G[li][n] = q_{l0+li} . Er[c0+n]
    int cc = c0 + colL;
    if (cc > LQ - 1) cc = LQ - 1;                // OOB cols are causally masked later
    v8f g = vz;
    #pragma unroll
    for (int f = 0; f < 2; ++f){
      FragA eb;
      const u16* p = er + (size_t)cc*64 + 32*f + half*16;
      eb.u[0] = *(const uint4*)p;
      eb.u[1] = *(const uint4*)(p + 8);
      g = wmma_bf16(aq[f].v, eb.v, g);
    }
    return g;
  };

  v8f carry = gtile(cs);                         // G tile j=0
  const int ntile = (l0 + 47) >> 5;              // key tiles of 32

  for (int t = 0; t < ntile; ++t){
    const int m0 = t << 5;
    if (t + 1 < ntile){                          // prefetch next key/value tiles
      __builtin_prefetch(k  + ((size_t)bh*LQ + m0 + 32 + colL)*64, 0, 0);
      __builtin_prefetch(vt + ((size_t)bh*64 + colL)*LQ + m0 + 32, 0, 0);
    }
    v8f g1 = gtile(cs + 32*t + 16);
    v8f g2 = gtile(cs + 32*t + 32);

    // S = Q @ K^T for 32 keys
    v8f s[2];
    #pragma unroll
    for (int ns = 0; ns < 2; ++ns){
      int key = m0 + ns*16 + colL;
      if (key > LQ - 1) key = LQ - 1;            // masked anyway
      const u16* krow = k + ((size_t)bh*LQ + key)*64;
      v8f sa = vz;
      #pragma unroll
      for (int f = 0; f < 2; ++f){
        FragA kf;
        const u16* p = krow + 32*f + half*16;
        kf.u[0] = *(const uint4*)p;
        kf.u[1] = *(const uint4*)(p + 8);
        sa = wmma_bf16(aq[f].v, kf.v, sa);
      }
      s[ns] = sa;
    }

    float alpha[8];
    #pragma unroll
    for (int i = 0; i < 8; ++i){
      const int li = i + 8*half;
      const int mi = colL;
      // Srel gather from Toeplitz band (C-layout: same VGPR slot, shuffled lane)
      int idx0 = 15 + mi - li;                   // [0..30] over {carry,g1}
      int src0 = half*16 + (idx0 & 15);
      float gc  = __shfl(carry[i], src0, 32);
      float g1a = __shfl(g1[i],    src0, 32);
      float srel0 = (idx0 & 16) ? g1a : gc;
      int j1 = idx0;                             // idx1-16 in [0..30] over {g1,g2}
      int src1 = half*16 + (j1 & 15);
      float g1b = __shfl(g1[i], src1, 32);
      float g2a = __shfl(g2[i], src1, 32);
      float srel1 = (j1 & 16) ? g2a : g1b;

      const int l_abs = l0 + li;
      float x0 = (m0 + mi      <= l_abs) ? (s[0][i] + srel0) * scale : -1e30f;
      float x1 = (m0 + 16 + mi <= l_abs) ? (s[1][i] + srel1) * scale : -1e30f;

      float rm = rmax16(fmaxf(x0, x1));
      float mn = fmaxf(m_i[i], rm);
      float al = __expf(m_i[i] - mn);
      float p0 = __expf(x0 - mn);
      float p1 = __expf(x1 - mn);
      float rs = rsum16(p0 + p1);
      l_i[i] = l_i[i]*al + rs;
      m_i[i] = mn;
      alpha[i] = al;
      pscr[li*40 + colL]      = f2bf(p0);
      pscr[li*40 + colL + 16] = f2bf(p1);
    }
    #pragma unroll
    for (int i = 0; i < 8; ++i){
      float al = alpha[i];
      o[0][i] *= al; o[1][i] *= al; o[2][i] *= al; o[3][i] *= al;
    }

    asm volatile("s_wait_dscnt 0" ::: "memory"); // P stores -> A-fragment loads (same wave)
    FragA pa;
    {
      const u16* pp = pscr + colL*40 + half*16;
      pa.u[0] = *(const uint4*)pp;
      pa.u[1] = *(const uint4*)(pp + 8);
    }
    // O += P @ V   (V transposed: vt[bh][d][l])
    #pragma unroll
    for (int ds = 0; ds < 4; ++ds){
      FragA vb;
      int d = ds*16 + colL;
      const u16* vp = vt + ((size_t)bh*64 + d)*LQ + m0 + half*16;
      vb.u[0] = *(const uint4*)vp;
      vb.u[1] = *(const uint4*)(vp + 8);
      o[ds] = wmma_bf16(pa.v, vb.v, o[ds]);
    }
    carry = g2;
  }

  // normalize + write y_att [b*2048+l][h*64+d] bf16
  const size_t bb = (size_t)(bh >> 4);
  const int h = bh & 15;
  #pragma unroll
  for (int i = 0; i < 8; ++i){
    const int li = i + 8*half;
    float inv = 1.0f / l_i[i];
    size_t rowbase = (bb*LQ + l0 + li)*1024 + (size_t)h*64;
    #pragma unroll
    for (int ds = 0; ds < 4; ++ds)
      yatt[rowbase + ds*16 + colL] = f2bf(o[ds][i] * inv);
  }
}

// ---------------- host ----------------

extern "C" void kernel_launch(void* const* d_in, const int* in_sizes, int n_in,
                              void* d_out, int out_size, void* d_ws, size_t ws_size,
                              hipStream_t stream)
{
  (void)in_sizes; (void)n_in; (void)out_size; (void)ws_size;
  const float* x      = (const float*)d_in[0];
  const float* W_attn = (const float*)d_in[1];
  const float* b_attn = (const float*)d_in[2];
  const float* Er     = (const float*)d_in[3];
  const float* W_proj = (const float*)d_in[4];
  const float* b_proj = (const float*)d_in[5];
  float* out = (float*)d_out;
  char* ws = (char*)d_ws;

  u16* xbf  = (u16*)(ws + (size_t)0);          // 8192*1024           (16 MB)
  u16* wat  = (u16*)(ws + (size_t)16777216);   // W_attn^T 3072*1024  ( 6 MB)
  u16* wpt  = (u16*)(ws + (size_t)23068672);   // W_proj^T 1024*1024  ( 2 MB)
  u16* erbf = (u16*)(ws + (size_t)25165824);   // Er 2048*64
  u16* qb   = (u16*)(ws + (size_t)25427968);   // q [bh][l][d]        (16 MB)
  u16* kb   = (u16*)(ws + (size_t)42205184);   // k [bh][l][d]        (16 MB)
  u16* vtb  = (u16*)(ws + (size_t)58982400);   // v^T [bh][d][l]      (16 MB)
  u16* yatt = (u16*)(ws + (size_t)75759616);   // attn out [tok][D]   (16 MB)

  cvt_bf16_kernel<<<32768, 256, 0, stream>>>(x, xbf, 8388608);
  cvt_bf16_kernel<<<512, 256, 0, stream>>>(Er, erbf, 131072);
  transpose_f32_bf16<<<dim3(96, 32), 256, 0, stream>>>(W_attn, wat, 1024, 3072);
  transpose_f32_bf16<<<dim3(32, 32), 256, 0, stream>>>(W_proj, wpt, 1024, 1024);

  gemm_bf16<<<dim3(64, 24), 256, 0, stream>>>(xbf, 1024, wat, 3072, b_attn, 0,
                                              qb, kb, vtb, nullptr);
  attn_kernel<<<2048, 128, 0, stream>>>(qb, kb, vtb, erbf, yatt);
  gemm_bf16<<<dim3(64, 8), 256, 0, stream>>>(yatt, 1024, wpt, 1024, b_proj, 1,
                                             nullptr, nullptr, nullptr, out);
}
